// _Model_51694226375203
// MI455X (gfx1250) — compile-verified, matrix operand order checked by reading, not run on hardware
//
#include <hip/hip_runtime.h>
#include <stdint.h>

// ---------------------------------------------------------------------------
// GCN forward on MI455X (gfx1250):
//   WMMA f16 GEMMs (TDM-staged A panels, pre-packed f16 B fragments)
//   + COO SpMM scatter-adds (layer-1 masked to idx destinations).
// ---------------------------------------------------------------------------

typedef __attribute__((ext_vector_type(16))) _Float16 v16h;
typedef __attribute__((ext_vector_type(8)))  float    v8f;
typedef __attribute__((ext_vector_type(4)))  float    v4f;
typedef unsigned int u32;
typedef __attribute__((ext_vector_type(4)))  u32      u32x4;
typedef __attribute__((ext_vector_type(8)))  u32      u32x8;

#define NNODES 50000
#define NEDGES 800000
#define F_IN   512
#define HID    128
#define OUTF   64
#define NIDX   1000

#define GCN_USE_TDM 1   // flip to 0 to fall back to plain LDS staging

// Generic -> LDS(addrspace 3) offset of a __shared__ object (bytes).
__device__ __forceinline__ u32 lds_offset_of(const void* p) {
    return (u32)(unsigned long long)(uintptr_t)
        ((__attribute__((address_space(3))) const void*)p);
}

// Issue TENSOR_LOAD_TO_LDS for a contiguous 1-D fp32 block (elems elements)
// from gaddr into LDS offset ldsOff. D# layout per CDNA5 ISA ch.8.
__device__ __forceinline__ void tdm_load_1d_f32(const void* gsrc, u32 ldsOff, u32 elems) {
    const unsigned long long ga = (unsigned long long)(uintptr_t)gsrc;
    u32x4 g0;
    g0[0] = 1u;                                    // count=1, user descriptor
    g0[1] = ldsOff;                                // lds_addr (bytes)
    g0[2] = (u32)(ga & 0xFFFFFFFFu);               // global_addr[31:0]
    g0[3] = (u32)((ga >> 32) & 0x1FFFFFFu)         // global_addr[56:32]
          | (2u << 30);                            // type = 2 ("image")
    u32x8 g1;
    g1[0] = (2u << 16);                            // data_size = 4 bytes
    g1[1] = (elems & 0xFFFFu) << 16;               // tensor_dim0[15:0]  @bits79:48
    g1[2] = ((elems >> 16) & 0xFFFFu)              // tensor_dim0[31:16]
          | (1u << 16);                            // tensor_dim1 = 1    @bits111:80
    g1[3] = (elems & 0xFFFFu) << 16;               // tile_dim0 = elems  @bits127:112
    g1[4] = 1u;                                    // tile_dim1 = 1 (tile_dim2 = 0)
    g1[5] = elems;                                 // tensor_dim0_stride[31:0]
    g1[6] = 0u;
    g1[7] = 0u;
    u32x4 gz = {0u, 0u, 0u, 0u};                   // groups 2/3 unused (2-D-or-less)
    asm volatile("tensor_load_to_lds %0, %1, %2, %3"
                 :: "s"(g0), "s"(g1), "s"(gz), "s"(gz)
                 : "memory");
}

// ---------------------------------------------------------------------------
// Pre-pack B[K x NCOLS] (fp32, row-major) into WMMA-fragment-ordered f16:
// thread t = ((ks * NCOLS/16 + ct) * 32 + lane) writes its 16 halves
// contiguously, so the GEMM reads one 32-byte chunk per lane per k-step.
// 16-bit B layout mirror of A: lane holds column n = ct*16 + (lane&15),
// K = ks*32 + (lane>>4)*8 + {0..7, 16..23}.
// ---------------------------------------------------------------------------
template <int K, int NCOLS>
__global__ void pack_b_f16(const float* __restrict__ B, _Float16* __restrict__ Bp) {
    constexpr int CT = NCOLS / 16;
    constexpr int TOTAL = (K / 32) * CT * 32;
    const int t = blockIdx.x * blockDim.x + threadIdx.x;
    if (t >= TOTAL) return;
    const int lane = t & 31;
    const int rest = t >> 5;
    const int ct   = rest % CT;
    const int ks   = rest / CT;
    const int base = ks * 32 + (lane >> 4) * 8;
    const int col  = ct * 16 + (lane & 15);
    _Float16* dst = Bp + (size_t)t * 16;
#pragma unroll
    for (int j = 0; j < 8; ++j) {
        dst[j]     = (_Float16)B[(size_t)(base + j) * NCOLS + col];
        dst[j + 8] = (_Float16)B[(size_t)(base + 16 + j) * NCOLS + col];
    }
}

// ---------------------------------------------------------------------------
// C[M x NCOLS] = A[M x K] @ B[K x NCOLS]; A fp32 (TDM-staged in LDS),
// B pre-packed f16 fragments, f32 accumulate via v_wmma_f32_16x16x32_f16.
// grid.x = M/16; block = (NCOLS/16) waves; each wave owns a 16x16 tile.
// ---------------------------------------------------------------------------
template <int K, int NCOLS>
__global__ void __launch_bounds__((NCOLS / 16) * 32)
gemm_wmma_f16(const float* __restrict__ A, const v16h* __restrict__ Bp,
              float* __restrict__ C) {
    __shared__ __attribute__((aligned(16))) float Apanel[16 * K];

    const int tid  = threadIdx.x;
    const int row0 = blockIdx.x * 16;
    const float* gsrc = A + (size_t)row0 * K;   // 16 x K panel is contiguous

#if GCN_USE_TDM
    if (tid < 32) {                             // one wave issues the DMA
        tdm_load_1d_f32(gsrc, lds_offset_of(Apanel), 16 * K);
    }
    __builtin_amdgcn_s_wait_tensorcnt(0);       // issuing wave drains TENSORcnt
    __syncthreads();                            // publish LDS to all waves
#else
    for (int i = tid; i < 16 * K / 4; i += blockDim.x)
        ((v4f*)Apanel)[i] = ((const v4f*)gsrc)[i];
    __syncthreads();
#endif

    const int wave = tid >> 5;
    const int lane = tid & 31;
    const int half = lane >> 4;
    const int l16  = lane & 15;

    const float* Arow = Apanel + l16 * K;
    v8f acc = {};

    for (int k0 = 0; k0 < K; k0 += 32) {
        // A fragment, 16-bit A 16x32 layout:
        //  lanes 0-15 : K = k0+{0..7},  k0+{16..23}
        //  lanes 16-31: K = k0+{8..15}, k0+{24..31}
        const int base = k0 + half * 8;
        const v4f x0 = *(const v4f*)(Arow + base);
        const v4f x1 = *(const v4f*)(Arow + base + 4);
        const v4f x2 = *(const v4f*)(Arow + base + 16);
        const v4f x3 = *(const v4f*)(Arow + base + 20);
        v16h a;
#pragma unroll
        for (int j = 0; j < 4; ++j) {
            a[j]      = (_Float16)x0[j];
            a[j + 4]  = (_Float16)x1[j];
            a[j + 8]  = (_Float16)x2[j];
            a[j + 12] = (_Float16)x3[j];
        }
        // B fragment: one contiguous 32-byte load.
        const v16h b = Bp[((size_t)(k0 >> 5) * (NCOLS / 16) + wave) * 32 + lane];

        acc = __builtin_amdgcn_wmma_f32_16x16x32_f16(
            false, a, false, b, (short)0, acc, false, false);
    }

    const int ncol0 = wave * 16;
#pragma unroll
    for (int r = 0; r < 8; ++r)
        C[(size_t)(row0 + r + half * 8) * NCOLS + ncol0 + l16] = acc[r];
}

// ---------------------------------------------------------------------------
// COO scatter: acc[dst] += w * msg[src]; one float4 chunk per thread.
// For D=128 a full wave covers one edge row (512B coalesced).
// ---------------------------------------------------------------------------
template <int D>
__global__ void spmm_scatter(const int* __restrict__ src,
                             const int* __restrict__ dst,
                             const float* __restrict__ w,
                             const float* __restrict__ msg,
                             float* __restrict__ acc) {
    constexpr int CH = D / 4;
    const long long t = (long long)blockIdx.x * blockDim.x + threadIdx.x;
    if (t >= (long long)NEDGES * CH) return;
    const int e = (int)(t / CH);
    const int c = (int)(t - (long long)e * CH) * 4;
    const int   s  = src[e];
    const int   d  = dst[e];
    const float wt = w[e];
    const v4f m = *(const v4f*)(msg + (size_t)s * D + c);
    float* p = acc + (size_t)d * D + c;
    atomicAdd(p + 0, wt * m[0]);
    atomicAdd(p + 1, wt * m[1]);
    atomicAdd(p + 2, wt * m[2]);
    atomicAdd(p + 3, wt * m[3]);
}

// Layer-1 variant: only edges whose destination is gathered at the end matter.
template <int D>
__global__ void spmm_scatter_masked(const int* __restrict__ src,
                                    const int* __restrict__ dst,
                                    const float* __restrict__ w,
                                    const float* __restrict__ msg,
                                    float* __restrict__ acc,
                                    const u32* __restrict__ flags) {
    constexpr int CH = D / 4;
    const long long t = (long long)blockIdx.x * blockDim.x + threadIdx.x;
    if (t >= (long long)NEDGES * CH) return;
    const int e = (int)(t / CH);
    const int d = dst[e];
    if (!flags[d]) return;                      // ~98% of edges drop out
    const int c = (int)(t - (long long)e * CH) * 4;
    const int   s  = src[e];
    const float wt = w[e];
    const v4f m = *(const v4f*)(msg + (size_t)s * D + c);
    float* p = acc + (size_t)d * D + c;
    atomicAdd(p + 0, wt * m[0]);
    atomicAdd(p + 1, wt * m[1]);
    atomicAdd(p + 2, wt * m[2]);
    atomicAdd(p + 3, wt * m[3]);
}

__global__ void zero_f32(float* __restrict__ p, int n) {
    const int i = blockIdx.x * blockDim.x + threadIdx.x;
    if (i < n) p[i] = 0.0f;
}
__global__ void zero_u32(u32* __restrict__ p, int n) {
    const int i = blockIdx.x * blockDim.x + threadIdx.x;
    if (i < n) p[i] = 0u;
}
__global__ void mark_idx(const int* __restrict__ idx, u32* __restrict__ flags) {
    const int i = blockIdx.x * blockDim.x + threadIdx.x;
    if (i < NIDX) flags[idx[i]] = 1u;
}
__global__ void relu_inplace(float* __restrict__ p, int n) {
    const int i = blockIdx.x * blockDim.x + threadIdx.x;
    if (i < n) p[i] = fmaxf(p[i], 0.0f);
}
__global__ void gather_rows(const float* __restrict__ full,
                            const int* __restrict__ idx,
                            float* __restrict__ out) {
    const int i = blockIdx.x * blockDim.x + threadIdx.x;
    if (i >= NIDX * OUTF) return;
    const int r = i / OUTF;
    const int c = i - r * OUTF;
    out[i] = full[(size_t)idx[r] * OUTF + c];
}

// ---------------------------------------------------------------------------
extern "C" void kernel_launch(void* const* d_in, const int* in_sizes, int n_in,
                              void* d_out, int out_size, void* d_ws, size_t ws_size,
                              hipStream_t stream) {
    const float* x   = (const float*)d_in[0];   // [N, 512]
    const float* W0  = (const float*)d_in[1];   // [512, 128]
    const float* W1  = (const float*)d_in[2];   // [128, 64]
    const float* ew  = (const float*)d_in[3];   // [E]
    const int*   esr = (const int*)d_in[4];     // [E]
    const int*   eds = (const int*)d_in[5];     // [E]
    const int*   idx = (const int*)d_in[6];     // [NIDX]
    float*       out = (float*)d_out;           // [NIDX, 64]

    const int nh = NNODES * HID;   // 6,400,000
    const int no = NNODES * OUTF;  // 3,200,000

    // Workspace layout (≈51.6 MB total):
    float*     reg0  = (float*)d_ws;                    // t0 / t1
    float*     reg1  = reg0 + (size_t)nh;               // hacc/h, later oacc
    u32*       flags = (u32*)(reg1 + (size_t)nh);       // [N] dst-needed mask
    _Float16*  w0p   = (_Float16*)(flags + NNODES);     // packed W0 (128 KB)
    _Float16*  w1p   = w0p + (size_t)F_IN * HID;        // packed W1 (16 KB)

    // 0) init: layer-0 accumulator, dst mask, packed weights
    zero_f32<<<(nh + 255) / 256, 256, 0, stream>>>(reg1, nh);
    zero_u32<<<(NNODES + 255) / 256, 256, 0, stream>>>(flags, NNODES);
    mark_idx<<<(NIDX + 255) / 256, 256, 0, stream>>>(idx, flags);
    pack_b_f16<F_IN, HID><<<((F_IN / 32) * (HID / 16) * 32 + 255) / 256, 256, 0, stream>>>(W0, w0p);
    pack_b_f16<HID, OUTF><<<((HID / 32) * (OUTF / 16) * 32 + 255) / 256, 256, 0, stream>>>(W1, w1p);

    // 1) t0 = x @ W0 (WMMA). 50000/16 = 3125 row panels, 8 waves/block.
    gemm_wmma_f16<F_IN, HID><<<NNODES / 16, (HID / 16) * 32, 0, stream>>>(
        x, (const v16h*)w0p, reg0);

    // 2) hacc[dst] += w * t0[src]
    {
        const long long tt = (long long)NEDGES * (HID / 4);
        spmm_scatter<HID><<<(int)((tt + 255) / 256), 256, 0, stream>>>(esr, eds, ew, reg0, reg1);
    }

    // 3) h = relu(hacc) in place
    relu_inplace<<<(nh + 255) / 256, 256, 0, stream>>>(reg1, nh);

    // 4) t1 = h @ W1 (WMMA). 4 waves/block.
    gemm_wmma_f16<HID, OUTF><<<NNODES / 16, (OUTF / 16) * 32, 0, stream>>>(
        reg1, (const v16h*)w1p, reg0);

    // 5) zero layer-1 accumulator (reuses reg1; h dead after GEMM2)
    zero_f32<<<(no + 255) / 256, 256, 0, stream>>>(reg1, no);

    // 6) oacc[dst] += w * t1[src], only for dst in idx (~2% of edges)
    {
        const long long tt = (long long)NEDGES * (OUTF / 4);
        spmm_scatter_masked<OUTF><<<(int)((tt + 255) / 256), 256, 0, stream>>>(
            esr, eds, ew, reg0, reg1, flags);
    }

    // 7) out = oacc[idx]
    gather_rows<<<(NIDX * OUTF + 255) / 256, 256, 0, stream>>>(reg1, idx, out);
}
